// Net_GCA_9096740733111
// MI455X (gfx1250) — matrix operand order repaired and checked by reading, not compiled, not run on hardware
//
#include <hip/hip_runtime.h>
#include <hip/hip_bf16.h>
#include <math.h>

#define BB 8
#define NN 2048
#define KK 20
#define C1 64
#define C2 256

typedef __attribute__((ext_vector_type(16))) __bf16 v16bf;
typedef __attribute__((ext_vector_type(8)))  __bf16 bf16x8;
typedef __attribute__((ext_vector_type(8)))  float  v8f;

// ---------------------------------------------------------------------------
// WMMA helpers (wave32, 16x16x32 bf16 -> f32)
// ---------------------------------------------------------------------------
__device__ __forceinline__ v8f wmma_bf16(v16bf a, v16bf b, v8f c) {
  return __builtin_amdgcn_wmma_f32_16x16x32_bf16(false, a, false, b, (short)0, c,
                                                 false, false);
}

// A tile 16x32 bf16, row-major source with leading dim `ld` (halves).
// Lane layout: m = m0 + (lane&15); K-base = k0 + (lane>=16 ? 8 : 0).
// Per lane the 16 halves are two contiguous 16B runs: [kb..kb+7], [kb+16..kb+23].
__device__ __forceinline__ v16bf pack_A_fast(const __bf16* src, int ld, int m0,
                                             int k0) {
  int lane = (int)threadIdx.x & 31;
  int m = m0 + (lane & 15);
  int kb = k0 + ((lane >> 4) ? 8 : 0);
  const __bf16* p = src + (size_t)m * ld + kb;
  union { v16bf v; bf16x8 h[2]; } u;
  u.h[0] = *(const bf16x8*)(p);
  u.h[1] = *(const bf16x8*)(p + 16);
  return u.v;
}

// B tile 32x16 bf16 from an N-major [Ndim][Kpad] source (i.e. B^T storage).
// Lane layout: n = n0 + (lane&15); K-base = k0 + (lane>=16 ? 8 : 0).
// Per lane the 16 halves are two contiguous 16B runs: [kb..kb+7], [kb+16..kb+23].
__device__ __forceinline__ v16bf pack_B_fast(const __bf16* srcT, int ldk, int k0,
                                             int n0) {
  int lane = (int)threadIdx.x & 31;
  int n = n0 + (lane & 15);
  int kb = k0 + ((lane >> 4) ? 8 : 0);
  const __bf16* p = srcT + (size_t)n * ldk + kb;
  union { v16bf v; bf16x8 h[2]; } u;
  u.h[0] = *(const bf16x8*)(p);
  u.h[1] = *(const bf16x8*)(p + 16);
  return u.v;
}

// ---------------------------------------------------------------------------
// conversion kernels
// ---------------------------------------------------------------------------
__global__ void k_f32_to_bf16(const float* __restrict__ in, __bf16* __restrict__ out,
                              int n) {
  int i = blockIdx.x * blockDim.x + threadIdx.x;
  if (i < n) out[i] = (__bf16)in[i];
}

// W [Kdim x Ndim] fp32 row-major -> Wt [Ndim x Kpad] bf16, zero padded K tail
__global__ void k_transpose_pad_bf16(const float* __restrict__ W,
                                     __bf16* __restrict__ out, int Kdim, int Ndim,
                                     int Kpad) {
  int t = blockIdx.x * blockDim.x + threadIdx.x;
  if (t >= Ndim * Kpad) return;
  int n = t / Kpad, k = t % Kpad;
  out[t] = (k < Kdim) ? (__bf16)W[(size_t)k * Ndim + n] : (__bf16)0.0f;
}

// ---------------------------------------------------------------------------
// top-K selection
// ---------------------------------------------------------------------------
__device__ void topk_select(float* d, float* rv, int* ri, int* out_idx) {
  int tid = (int)threadIdx.x;
  for (int t = 0; t < KK; ++t) {
    float best = 3.0e38f;
    int bi = 0;
    for (int j = tid; j < NN; j += 256) {
      float v = d[j];
      if (v < best) { best = v; bi = j; }
    }
    rv[tid] = best; ri[tid] = bi;
    __syncthreads();
    for (int s = 128; s > 0; s >>= 1) {
      if (tid < s) {
        float a = rv[tid], b = rv[tid + s];
        if (b < a || (b == a && ri[tid + s] < ri[tid])) {
          rv[tid] = b; ri[tid] = ri[tid + s];
        }
      }
      __syncthreads();
    }
    if (tid == 0) {
      out_idx[t] = ri[0];
      d[ri[0]] = 3.0e38f;
    }
    __syncthreads();
  }
}

__global__ __launch_bounds__(256) void k_knn6(const float* __restrict__ x,
                                              int* __restrict__ idx) {
  int row = blockIdx.x;
  int b = row / NN, i = row % NN;
  __shared__ float d[NN];
  __shared__ float rv[256];
  __shared__ int ri[256];
  const float* xb = x + (size_t)b * NN * 6;
  float xi[6];
#pragma unroll
  for (int f = 0; f < 6; ++f) xi[f] = xb[(size_t)i * 6 + f];
  for (int j = (int)threadIdx.x; j < NN; j += 256) {
    float s = 0.f;
#pragma unroll
    for (int f = 0; f < 6; ++f) {
      float t = xi[f] - xb[(size_t)j * 6 + f];
      s += t * t;
    }
    d[j] = s;
  }
  __syncthreads();
  topk_select(d, rv, ri, idx + (size_t)row * KK);
}

// 64-d pairwise distances via WMMA Gram, one wave per 16x64 output strip
__global__ __launch_bounds__(128) void k_gram(const __bf16* __restrict__ hb,
                                              const float* __restrict__ sq,
                                              float* __restrict__ D, int b) {
  int w = blockIdx.x * 4 + ((int)threadIdx.x >> 5);  // 0..4095
  int I = w >> 5;        // 0..127 (16-row strip)
  int Jg = w & 31;       // 0..31  (4 J-tiles each)
  int lane = (int)threadIdx.x & 31;
  const __bf16* base = hb + (size_t)b * NN * C1;
  v16bf a0 = pack_A_fast(base, C1, I * 16, 0);
  v16bf a1 = pack_A_fast(base, C1, I * 16, 32);
  int hi8 = (lane >> 4) ? 8 : 0;
  float sqr[8];
#pragma unroll
  for (int r = 0; r < 8; ++r) sqr[r] = sq[(size_t)b * NN + I * 16 + r + hi8];
#pragma unroll
  for (int jt = 0; jt < 4; ++jt) {
    int J = Jg * 4 + jt;
    v16bf b0 = pack_B_fast(base, C1, 0, J * 16);   // h1b rows as columns
    v16bf b1 = pack_B_fast(base, C1, 32, J * 16);
    v8f acc = {};
    acc = wmma_bf16(a0, b0, acc);
    acc = wmma_bf16(a1, b1, acc);
    int col = J * 16 + (lane & 15);
    float sqj = sq[(size_t)b * NN + col];
#pragma unroll
    for (int r = 0; r < 8; ++r) {
      int row = I * 16 + r + hi8;
      D[(size_t)row * NN + col] = sqr[r] + sqj - 2.0f * acc[r];
    }
  }
}

__global__ __launch_bounds__(256) void k_topkD(const float* __restrict__ D,
                                               int* __restrict__ idx, int b) {
  int i = blockIdx.x;
  __shared__ float d[NN];
  __shared__ float rv[256];
  __shared__ int ri[256];
  for (int j = (int)threadIdx.x; j < NN; j += 256) d[j] = D[(size_t)i * NN + j];
  __syncthreads();
  topk_select(d, rv, ri, idx + ((size_t)b * NN + i) * KK);
}

// ---------------------------------------------------------------------------
// pos nets:  relu((h W + b)*s + t)
// ---------------------------------------------------------------------------
__global__ __launch_bounds__(64) void k_pos1(
    const float* __restrict__ x, const float* W1, const float* b1, const float* s1,
    const float* t1, const float* W2, const float* b2, const float* s2,
    const float* t2, float* __restrict__ p1) {
  int pt = blockIdx.x;
  int b = pt / NN, i = pt % NN;
  int tid = (int)threadIdx.x;
  __shared__ float pp[3];
  __shared__ float hid[32];
  if (tid < 3) pp[tid] = x[((size_t)b * NN + i) * 6 + tid];
  __syncthreads();
  if (tid < 32) {
    float a = b1[tid];
#pragma unroll
    for (int f = 0; f < 3; ++f) a += pp[f] * W1[f * 32 + tid];
    a = a * s1[tid] + t1[tid];
    hid[tid] = fmaxf(a, 0.f);
  }
  __syncthreads();
  float a = b2[tid];
  for (int k = 0; k < 32; ++k) a += hid[k] * W2[k * 64 + tid];
  a = a * s2[tid] + t2[tid];
  p1[((size_t)b * NN + i) * C1 + tid] = fmaxf(a, 0.f);
}

__global__ __launch_bounds__(256) void k_pos2(
    const float* __restrict__ x, const float* W1, const float* b1, const float* s1,
    const float* t1, const float* W2, const float* b2, const float* s2,
    const float* t2, float* __restrict__ p2) {
  int pt = blockIdx.x;
  int b = pt / NN, i = pt % NN;
  int tid = (int)threadIdx.x;
  __shared__ float pp[3];
  __shared__ float hid[128];
  if (tid < 3) pp[tid] = x[((size_t)b * NN + i) * 6 + tid];
  __syncthreads();
  if (tid < 128) {
    float a = b1[tid];
#pragma unroll
    for (int f = 0; f < 3; ++f) a += pp[f] * W1[f * 128 + tid];
    a = a * s1[tid] + t1[tid];
    hid[tid] = fmaxf(a, 0.f);
  }
  __syncthreads();
  float a = b2[tid];
  for (int k = 0; k < 128; ++k) a += hid[k] * W2[k * 256 + tid];
  a = a * s2[tid] + t2[tid];
  p2[((size_t)b * NN + i) * C2 + tid] = fmaxf(a, 0.f);
}

// ---------------------------------------------------------------------------
// EdgeConv layers.  Rows >= KK carry garbage through the matmuls (rows are
// independent) and are masked only at the final max reduction.
// epilogue: relu(acc + bias)*scale + shift
// ---------------------------------------------------------------------------
template <int KSTEPS>
__device__ void edge1_layer(const __bf16 (*in)[C1], __bf16 (*out)[C1],
                            const __bf16* Wt /* [64][KSTEPS*32] */,
                            const float* bias, const float* sc, const float* sh) {
  int lane = (int)threadIdx.x & 31;
  v16bf A[2][KSTEPS];
#pragma unroll
  for (int mt = 0; mt < 2; ++mt)
#pragma unroll
    for (int ks = 0; ks < KSTEPS; ++ks)
      A[mt][ks] = pack_A_fast(&in[0][0], C1, mt * 16, ks * 32);
  int hi8 = (lane >> 4) ? 8 : 0;
#pragma unroll
  for (int nt = 0; nt < 4; ++nt) {
    v8f acc0 = {}, acc1 = {};
#pragma unroll
    for (int ks = 0; ks < KSTEPS; ++ks) {
      v16bf Bt = pack_B_fast(Wt, KSTEPS * 32, ks * 32, nt * 16);
      acc0 = wmma_bf16(A[0][ks], Bt, acc0);
      acc1 = wmma_bf16(A[1][ks], Bt, acc1);
    }
    int n = nt * 16 + (lane & 15);
    float bn = bias[n], sn = sc[n], tn = sh[n];
#pragma unroll
    for (int r = 0; r < 8; ++r) {
      out[r + hi8][n] = (__bf16)(fmaxf(acc0[r] + bn, 0.f) * sn + tn);
      out[16 + r + hi8][n] = (__bf16)(fmaxf(acc1[r] + bn, 0.f) * sn + tn);
    }
  }
}

__global__ __launch_bounds__(128) void k_edge1(
    const float* __restrict__ x, const int* __restrict__ idx,
    const float* __restrict__ p1, const __bf16* W1t, const float* bb1,
    const float* ss1, const float* tt1, const __bf16* W2t, const float* bb2,
    const float* ss2, const float* tt2, const __bf16* W3t, const float* bb3,
    const float* ss3, const float* tt3, float* __restrict__ h1) {
  int wave = (int)threadIdx.x >> 5, lane = (int)threadIdx.x & 31;
  int pt = blockIdx.x * 4 + wave;
  int b = pt / NN, i = pt % NN;
  __shared__ __bf16 bufA[4][32][C1];
  __shared__ __bf16 bufB[4][32][C1];
  __bf16(*cur)[C1] = bufA[wave];
  __bf16(*nxt)[C1] = bufB[wave];
  const float* xb = x + (size_t)b * NN * 6;
  // layer-1 reads K cols 0..31: zero the pad (cols 12..31) on every row;
  // rows >= KK zeroed entirely (cheap, avoids NaN garbage).
#pragma unroll
  for (int f = 12; f < 32; ++f) cur[lane][f] = (__bf16)0.0f;
  if (lane < KK) {
    int j = idx[((size_t)b * NN + i) * KK + lane];
#pragma unroll
    for (int f = 0; f < 6; ++f) {
      float xi = xb[(size_t)i * 6 + f];
      float xj = xb[(size_t)j * 6 + f];
      cur[lane][f] = (__bf16)xi;
      cur[lane][6 + f] = (__bf16)(xj - xi);
    }
  } else {
#pragma unroll
    for (int f = 0; f < 12; ++f) cur[lane][f] = (__bf16)0.0f;
  }
  __syncthreads();
  edge1_layer<1>(cur, nxt, W1t, bb1, ss1, tt1);
  __syncthreads();
  edge1_layer<2>(nxt, cur, W2t, bb2, ss2, tt2);
  __syncthreads();
  edge1_layer<2>(cur, nxt, W3t, bb3, ss3, tt3);
  __syncthreads();
#pragma unroll
  for (int cc = 0; cc < 2; ++cc) {
    int c = lane + cc * 32;
    float mx = -3.0e38f;
    for (int r = 0; r < KK; ++r) mx = fmaxf(mx, (float)nxt[r][c]);
    size_t o = ((size_t)b * NN + i) * C1 + c;
    h1[o] = mx + p1[o];
  }
}

__global__ __launch_bounds__(128) void k_edge2(
    const float* __restrict__ h1, const int* __restrict__ idx,
    const float* __restrict__ p2, const __bf16* Wt /* [256][128] */,
    const float* bias, const float* sc, const float* sh, float* __restrict__ h2) {
  int wave = (int)threadIdx.x >> 5, lane = (int)threadIdx.x & 31;
  int pt = blockIdx.x * 4 + wave;
  int b = pt / NN, i = pt % NN;
  __shared__ __bf16 feat[4][32][128];
  __bf16(*F)[128] = feat[wave];
  __builtin_prefetch(Wt, 0, 3);  // global_prefetch_b8 of the weight block
  const float* hb = h1 + (size_t)b * NN * C1;
  if (lane < KK) {
    int j = idx[((size_t)b * NN + i) * KK + lane];
    for (int f = 0; f < C1; ++f) {
      float hi = hb[(size_t)i * C1 + f];
      float hj = hb[(size_t)j * C1 + f];
      F[lane][f] = (__bf16)hi;
      F[lane][64 + f] = (__bf16)(hj - hi);
    }
  } else {
    for (int f = 0; f < 128; ++f) F[lane][f] = (__bf16)0.0f;
  }
  __syncthreads();
  v16bf A[2][4];
#pragma unroll
  for (int mt = 0; mt < 2; ++mt)
#pragma unroll
    for (int ks = 0; ks < 4; ++ks)
      A[mt][ks] = pack_A_fast(&F[0][0], 128, mt * 16, ks * 32);
  int hi8 = (lane >> 4) ? 8 : 0;
#pragma unroll
  for (int nt = 0; nt < 16; ++nt) {
    v8f acc0 = {}, acc1 = {};
#pragma unroll
    for (int ks = 0; ks < 4; ++ks) {
      v16bf Bt = pack_B_fast(Wt, 128, ks * 32, nt * 16);
      acc0 = wmma_bf16(A[0][ks], Bt, acc0);
      acc1 = wmma_bf16(A[1][ks], Bt, acc1);
    }
    int n = nt * 16 + (lane & 15);
    float bn = bias[n], sn = sc[n], tn = sh[n];
    float mx = -3.0e38f;
#pragma unroll
    for (int r = 0; r < 8; ++r) {
      int row0 = r + hi8;
      float v0 = fmaxf(acc0[r] + bn, 0.f) * sn + tn;
      if (row0 < KK) mx = fmaxf(mx, v0);
      int row1 = 16 + r + hi8;
      float v1 = fmaxf(acc1[r] + bn, 0.f) * sn + tn;
      if (row1 < KK) mx = fmaxf(mx, v1);
    }
    mx = fmaxf(mx, __shfl_xor(mx, 16, 32));
    if (lane < 16) {
      size_t o = ((size_t)b * NN + i) * C2 + n;
      h2[o] = mx + p2[o];
    }
  }
}

// ---------------------------------------------------------------------------
// SE blocks, pooling, classifier
// ---------------------------------------------------------------------------
__global__ void k_colmean(const float* __restrict__ h, float* __restrict__ avg,
                          int C) {
  int t = blockIdx.x * blockDim.x + threadIdx.x;
  if (t >= BB * C) return;
  int b = t / C, c = t % C;
  const float* p = h + (size_t)b * NN * C + c;
  float s = 0.f;
  for (int n = 0; n < NN; ++n) s += p[(size_t)n * C];
  avg[t] = s * (1.0f / NN);
}

__global__ void k_se(const float* __restrict__ avg, const float* W1,
                     const float* b1, const float* W2, const float* b2,
                     float* __restrict__ attn, int C, int H) {
  int b = blockIdx.x;
  extern __shared__ float sm[];
  float* av = sm;
  float* hid = sm + C;
  for (int c = (int)threadIdx.x; c < C; c += blockDim.x) av[c] = avg[b * C + c];
  __syncthreads();
  for (int h = (int)threadIdx.x; h < H; h += blockDim.x) {
    float a = b1[h];
    for (int c = 0; c < C; ++c) a += av[c] * W1[c * H + h];
    hid[h] = fmaxf(a, 0.f);
  }
  __syncthreads();
  for (int c = (int)threadIdx.x; c < C; c += blockDim.x) {
    float a = b2[c];
    for (int h = 0; h < H; ++h) a += hid[h] * W2[h * C + c];
    attn[b * C + c] = 1.0f / (1.0f + expf(-a));
  }
}

__global__ void k_scale(float* __restrict__ h, const float* __restrict__ attn,
                        int C, int total) {
  int t = blockIdx.x * blockDim.x + threadIdx.x;
  if (t >= total) return;
  int c = t % C;
  int b = t / (NN * C);
  h[t] *= attn[b * C + c];
}

__global__ void k_sqnorm(const float* __restrict__ h1, float* __restrict__ sq) {
  int t = blockIdx.x * blockDim.x + threadIdx.x;
  if (t >= BB * NN) return;
  const float* p = h1 + (size_t)t * C1;
  float s = 0.f;
  for (int c = 0; c < C1; ++c) s += p[c] * p[c];
  sq[t] = s;
}

__global__ void k_maxpool(const float* __restrict__ h2, float* __restrict__ gmax) {
  int t = blockIdx.x * blockDim.x + threadIdx.x;
  if (t >= BB * C2) return;
  int b = t / C2, c = t % C2;
  const float* p = h2 + (size_t)b * NN * C2 + c;
  float m = -3.0e38f;
  for (int n = 0; n < NN; ++n) m = fmaxf(m, p[(size_t)n * C2]);
  gmax[t] = m;
}

__global__ void k_scalevec(const float* __restrict__ gmax,
                           const float* __restrict__ attn,
                           float* __restrict__ gvec) {
  int t = blockIdx.x * blockDim.x + threadIdx.x;
  if (t >= BB * C2) return;
  gvec[t] = gmax[t] * attn[t];  // sigmoid>0 => max commutes with channel scale
}

__global__ __launch_bounds__(64) void k_cls(
    const float* __restrict__ g1, const float* __restrict__ g2, const float* W,
    const float* b, const float* s, const float* t, const float* Wc,
    const float* bc, float* __restrict__ out) {
  int bi = blockIdx.x;
  int tid = (int)threadIdx.x;
  __shared__ float d[C2];
  __shared__ float h[64];
  __shared__ float lg[5];
  for (int c = tid; c < C2; c += 64) d[c] = g2[bi * C2 + c] - g1[bi * C2 + c];
  __syncthreads();
  float a = b[tid];
  for (int k = 0; k < C2; ++k) a += d[k] * W[k * 64 + tid];
  h[tid] = fmaxf(a, 0.f) * s[tid] + t[tid];
  __syncthreads();
  if (tid < 5) {
    float a2 = bc[tid];
    for (int k = 0; k < 64; ++k) a2 += h[k] * Wc[k * 5 + tid];
    lg[tid] = a2;
  }
  __syncthreads();
  if (tid == 0) {
    float m = -3.0e38f;
    for (int k = 0; k < 5; ++k) m = fmaxf(m, lg[k]);
    float se = 0.f;
    for (int k = 0; k < 5; ++k) se += expf(lg[k] - m);
    float lse = m + logf(se);
    for (int k = 0; k < 5; ++k) out[bi * 5 + k] = lg[k] - lse;
  }
}

// ---------------------------------------------------------------------------
// host side
// ---------------------------------------------------------------------------
struct NetPtrs {
  const float *c1W[3], *c1b[3], *c1s[3], *c1t[3];
  const float *c2W, *c2b, *c2s, *c2t;
  const float *se1w1, *se1b1, *se1w2, *se1b2;
  const float *se2w1, *se2b1, *se2w2, *se2b2;
  const float *p1W[2], *p1b[2], *p1s[2], *p1t[2];
  const float *p2W[2], *p2b[2], *p2s[2], *p2t[2];
  int* idx;
  float *h1, *p1buf, *p2buf, *h2, *sq, *D;
  float *avg1, *attn1, *avg2, *attn2, *gmax;
  __bf16 *h1b, *w1t, *w2t, *w3t, *wc2t;
};

static void run_branch(const NetPtrs& P, const float* xin, float* gvec,
                       hipStream_t stream) {
  k_knn6<<<BB * NN, 256, 0, stream>>>(xin, P.idx);
  k_pos1<<<BB * NN, 64, 0, stream>>>(xin, P.p1W[0], P.p1b[0], P.p1s[0], P.p1t[0],
                                     P.p1W[1], P.p1b[1], P.p1s[1], P.p1t[1],
                                     P.p1buf);
  k_edge1<<<BB * NN / 4, 128, 0, stream>>>(
      xin, P.idx, P.p1buf, P.w1t, P.c1b[0], P.c1s[0], P.c1t[0], P.w2t, P.c1b[1],
      P.c1s[1], P.c1t[1], P.w3t, P.c1b[2], P.c1s[2], P.c1t[2], P.h1);
  k_colmean<<<(BB * C1 + 255) / 256, 256, 0, stream>>>(P.h1, P.avg1, C1);
  k_se<<<BB, 256, (C1 + 16) * sizeof(float), stream>>>(P.avg1, P.se1w1, P.se1b1,
                                                       P.se1w2, P.se1b2, P.attn1,
                                                       C1, 16);
  {
    int total = BB * NN * C1;
    k_scale<<<(total + 255) / 256, 256, 0, stream>>>(P.h1, P.attn1, C1, total);
  }
  k_sqnorm<<<(BB * NN + 255) / 256, 256, 0, stream>>>(P.h1, P.sq);
  {
    int n = BB * NN * C1;
    k_f32_to_bf16<<<(n + 255) / 256, 256, 0, stream>>>(P.h1, P.h1b, n);
  }
  for (int b = 0; b < BB; ++b) {
    k_gram<<<1024, 128, 0, stream>>>(P.h1b, P.sq, P.D, b);
    k_topkD<<<NN, 256, 0, stream>>>(P.D, P.idx, b);
  }
  k_pos2<<<BB * NN, 256, 0, stream>>>(xin, P.p2W[0], P.p2b[0], P.p2s[0], P.p2t[0],
                                      P.p2W[1], P.p2b[1], P.p2s[1], P.p2t[1],
                                      P.p2buf);
  k_edge2<<<BB * NN / 4, 128, 0, stream>>>(P.h1, P.idx, P.p2buf, P.wc2t, P.c2b,
                                           P.c2s, P.c2t, P.h2);
  k_colmean<<<(BB * C2 + 255) / 256, 256, 0, stream>>>(P.h2, P.avg2, C2);
  k_se<<<BB, 256, (C2 + 64) * sizeof(float), stream>>>(P.avg2, P.se2w1, P.se2b1,
                                                       P.se2w2, P.se2b2, P.attn2,
                                                       C2, 64);
  k_maxpool<<<(BB * C2 + 255) / 256, 256, 0, stream>>>(P.h2, P.gmax);
  k_scalevec<<<(BB * C2 + 255) / 256, 256, 0, stream>>>(P.gmax, P.attn2, gvec);
}

extern "C" void kernel_launch(void* const* d_in, const int* in_sizes, int n_in,
                              void* d_out, int out_size, void* d_ws, size_t ws_size,
                              hipStream_t stream) {
  (void)in_sizes; (void)n_in; (void)out_size; (void)ws_size;
  const float* x = (const float*)d_in[0];
  const float* x2 = (const float*)d_in[1];
  NetPtrs P;
  int q = 2;
  for (int l = 0; l < 3; ++l) {
    P.c1W[l] = (const float*)d_in[q++]; P.c1b[l] = (const float*)d_in[q++];
    P.c1s[l] = (const float*)d_in[q++]; P.c1t[l] = (const float*)d_in[q++];
  }
  P.c2W = (const float*)d_in[q++]; P.c2b = (const float*)d_in[q++];
  P.c2s = (const float*)d_in[q++]; P.c2t = (const float*)d_in[q++];
  P.se1w1 = (const float*)d_in[q++]; P.se1b1 = (const float*)d_in[q++];
  P.se1w2 = (const float*)d_in[q++]; P.se1b2 = (const float*)d_in[q++];
  P.se2w1 = (const float*)d_in[q++]; P.se2b1 = (const float*)d_in[q++];
  P.se2w2 = (const float*)d_in[q++]; P.se2b2 = (const float*)d_in[q++];
  for (int l = 0; l < 2; ++l) {
    P.p1W[l] = (const float*)d_in[q++]; P.p1b[l] = (const float*)d_in[q++];
    P.p1s[l] = (const float*)d_in[q++]; P.p1t[l] = (const float*)d_in[q++];
  }
  for (int l = 0; l < 2; ++l) {
    P.p2W[l] = (const float*)d_in[q++]; P.p2b[l] = (const float*)d_in[q++];
    P.p2s[l] = (const float*)d_in[q++]; P.p2t[l] = (const float*)d_in[q++];
  }
  const float* mW = (const float*)d_in[q++];
  const float* mb = (const float*)d_in[q++];
  const float* ms = (const float*)d_in[q++];
  const float* mt = (const float*)d_in[q++];
  const float* Wc = (const float*)d_in[q++];
  const float* bc = (const float*)d_in[q++];

  char* ws = (char*)d_ws;
  size_t off = 0;
  auto take = [&](size_t bytes) {
    char* p = ws + off;
    off += (bytes + 255) & ~(size_t)255;
    return p;
  };
  P.idx   = (int*)take((size_t)BB * NN * KK * 4);
  P.h1    = (float*)take((size_t)BB * NN * C1 * 4);
  P.p1buf = (float*)take((size_t)BB * NN * C1 * 4);
  P.p2buf = (float*)take((size_t)BB * NN * C2 * 4);
  P.h2    = (float*)take((size_t)BB * NN * C2 * 4);
  P.D     = (float*)take((size_t)NN * NN * 4);
  P.sq    = (float*)take((size_t)BB * NN * 4);
  P.h1b   = (__bf16*)take((size_t)BB * NN * C1 * 2);
  P.avg1  = (float*)take(BB * C1 * 4);
  P.attn1 = (float*)take(BB * C1 * 4);
  P.avg2  = (float*)take(BB * C2 * 4);
  P.attn2 = (float*)take(BB * C2 * 4);
  P.gmax  = (float*)take(BB * C2 * 4);
  float* gvec1 = (float*)take(BB * C2 * 4);
  float* gvec2 = (float*)take(BB * C2 * 4);
  P.w1t  = (__bf16*)take((size_t)64 * 32 * 2);    // [64][32]  (K=12 padded)
  P.w2t  = (__bf16*)take((size_t)64 * 64 * 2);    // [64][64]
  P.w3t  = (__bf16*)take((size_t)64 * 64 * 2);    // [64][64]
  P.wc2t = (__bf16*)take((size_t)256 * 128 * 2);  // [256][128]

  // transpose+pad weights to N-major bf16 (contiguous WMMA-B per lane)
  k_transpose_pad_bf16<<<(64 * 32 + 255) / 256, 256, 0, stream>>>(P.c1W[0], P.w1t,
                                                                  12, 64, 32);
  k_transpose_pad_bf16<<<(64 * 64 + 255) / 256, 256, 0, stream>>>(P.c1W[1], P.w2t,
                                                                  64, 64, 64);
  k_transpose_pad_bf16<<<(64 * 64 + 255) / 256, 256, 0, stream>>>(P.c1W[2], P.w3t,
                                                                  64, 64, 64);
  k_transpose_pad_bf16<<<(256 * 128 + 255) / 256, 256, 0, stream>>>(
      P.c2W, P.wc2t, 128, 256, 128);

  run_branch(P, x, gvec1, stream);
  run_branch(P, x2, gvec2, stream);

  k_cls<<<BB, 64, 0, stream>>>(gvec1, gvec2, mW, mb, ms, mt, Wc, bc,
                               (float*)d_out);
}
